// AutoEncoder_53781580481200
// MI455X (gfx1250) — matrix operand order repaired and checked by reading, not compiled, not run on hardware
//
#include <hip/hip_runtime.h>
#include <hip/hip_bf16.h>

typedef _Float16 v16h __attribute__((ext_vector_type(16)));
typedef _Float16 v8h  __attribute__((ext_vector_type(8)));
typedef float    v8f  __attribute__((ext_vector_type(8)));
typedef float    v4f  __attribute__((ext_vector_type(4)));

#define ATOMS_N   262144
#define DIM       128
#define NSYM      4

// ---- workspace byte offsets ----
#define OFF_CNT   0                         // int[4]  per-symbol counts
#define OFF_CUR   16                        // int[4]  scatter cursors
#define OFF_TS    32                        // int[5]  tileStart (tile index per expert)
#define OFF_RB    64                        // int[4]  rowBase (padded row base per expert)
#define OFF_PERM  1024                      // int[N+512] routed atom indices (-1 = pad)
#define PERM_LEN  (ATOMS_N + 512)
#define OFF_EATOM (OFF_PERM + PERM_LEN * 4)             // float[N]
#define OFF_PW1   (OFF_EATOM + ATOMS_N * 4)             // f16, S*128*128
#define OFF_PW2   (OFF_PW1 + NSYM * DIM * DIM * 2)      // f16, S*128*128
#define WS_NEEDED (OFF_PW2 + NSYM * DIM * DIM * 2)

// ---------------- routing kernels ----------------

__global__ void k_zero_counters(int* ws_ints) {
    if (threadIdx.x < 8) ws_ints[threadIdx.x] = 0;   // cnt[4] + cur[4]
}

__global__ __launch_bounds__(256) void k_hist(const int* __restrict__ sym,
                                              int* __restrict__ cnt, int n) {
    __shared__ int lc[NSYM];
    int t = threadIdx.x;
    if (t < NSYM) lc[t] = 0;
    __syncthreads();
    int i = blockIdx.x * 256 + t;
    if (i < n) atomicAdd(&lc[sym[i]], 1);
    __syncthreads();
    if (t < NSYM && lc[t]) atomicAdd(&cnt[t], lc[t]);
}

__global__ void k_setup_offsets(const int* __restrict__ cnt,
                                int* __restrict__ tileStart,
                                int* __restrict__ rowBase,
                                int* __restrict__ perm) {
    if (threadIdx.x == 0 && blockIdx.x == 0) {
        int base = 0;
        for (int s = 0; s < NSYM; ++s) {
            tileStart[s] = base / 128;
            rowBase[s]   = base;
            int c = cnt[s];
            int padded = ((c + 127) / 128) * 128;
            for (int i = c; i < padded; ++i) perm[base + i] = -1;
            base += padded;
        }
        tileStart[NSYM] = base / 128;
    }
}

__global__ __launch_bounds__(256) void k_scatter(const int* __restrict__ sym,
                                                 const int* __restrict__ rowBase,
                                                 int* __restrict__ cur,
                                                 int* __restrict__ perm, int n) {
    __shared__ int lc[NSYM];
    __shared__ int lbase[NSYM];
    int t = threadIdx.x;
    if (t < NSYM) lc[t] = 0;
    __syncthreads();
    int i = blockIdx.x * 256 + t;
    int s = 0, myoff = 0;
    if (i < n) { s = sym[i]; myoff = atomicAdd(&lc[s], 1); }
    __syncthreads();
    if (t < NSYM) lbase[t] = lc[t] ? atomicAdd(&cur[t], lc[t]) : 0;
    __syncthreads();
    if (i < n) perm[rowBase[s] + lbase[s] + myoff] = i;
}

// ---------------- weight pre-pack into WMMA B-fragment order ----------------
// pW[((s*4+ks)*8+nt)*512 + lane*16 + i] = (f16) W[s][ ks*32 + 8*(lane>>4) + (i&7) + 16*(i>>3) ][ nt*16 + (lane&15) ]
__global__ __launch_bounds__(256) void k_pack_weights(const float* __restrict__ W1,
                                                      const float* __restrict__ W2,
                                                      _Float16* __restrict__ pW1,
                                                      _Float16* __restrict__ pW2) {
    int g = blockIdx.x * 256 + threadIdx.x;      // 0 .. 2*65536-1
    int layer = g >> 16;
    int li = g & 65535;
    int i  = li & 15;
    int l  = (li >> 4) & 31;
    int nt = (li >> 9) & 7;
    int ks = (li >> 12) & 3;
    int s  = (li >> 14) & 3;
    int k = ks * 32 + (l >> 4) * 8 + (i & 7) + (i >> 3) * 16;
    int n = nt * 16 + (l & 15);
    const float* W = layer ? W2 : W1;
    _Float16*    P = layer ? pW2 : pW1;
    P[li] = (_Float16)W[s * (DIM * DIM) + k * DIM + n];
}

// ---------------- main routed dual-GEMM kernel ----------------

__global__ __launch_bounds__(256) void k_moe_wmma(
    const float* __restrict__ x,
    const float* __restrict__ b1, const float* __restrict__ b2,
    const float* __restrict__ W3, const float* __restrict__ b3,
    const float* __restrict__ slope, const float* __restrict__ intercept,
    const _Float16* __restrict__ pW1, const _Float16* __restrict__ pW2,
    const int* __restrict__ perm, const int* __restrict__ tileStart,
    float* __restrict__ e_atom) {
    __shared__ __align__(16) _Float16 xT[128 * 128];   // 32 KB f16 input tile
    __shared__ __align__(16) _Float16 hT[128 * 128];   // 32 KB f16 hidden tile
    __shared__ __align__(32) _Float16 wL1[128 * 128];  // 32 KB expert W1 fragments
    __shared__ __align__(32) _Float16 wL2[128 * 128];  // 32 KB expert W2 fragments

    const int b = blockIdx.x;
    if (b >= tileStart[4]) return;
    int s = 0;
    if (b >= tileStart[1]) s = 1;
    if (b >= tileStart[2]) s = 2;
    if (b >= tileStart[3]) s = 3;

    const int tid = threadIdx.x;

    // ---- async copy this expert's packed weights into LDS (ASYNCcnt path) ----
    {
        const _Float16* gw1 = pW1 + (size_t)s * (DIM * DIM);
        const _Float16* gw2 = pW2 + (size_t)s * (DIM * DIM);
        // low 32 bits of the generic pointer to a __shared__ object == LDS offset
        unsigned l1off = (unsigned)(size_t)(const void*)&wL1[0];
        unsigned l2off = (unsigned)(size_t)(const void*)&wL2[0];
        #pragma unroll
        for (int it = 0; it < 8; ++it) {
            int e = it * 256 + tid;                 // 16-byte chunk index, 0..2047
            const void* g1 = (const void*)(gw1 + e * 8);
            const void* g2 = (const void*)(gw2 + e * 8);
            unsigned d1 = l1off + e * 16;
            unsigned d2 = l2off + e * 16;
            asm volatile("global_load_async_to_lds_b128 %0, %1, off"
                         :: "v"(d1), "v"(g1) : "memory");
            asm volatile("global_load_async_to_lds_b128 %0, %1, off"
                         :: "v"(d2), "v"(g2) : "memory");
        }
    }

    // ---- gather 128 atom rows -> f16 LDS tile (2 threads per row), overlapped ----
    {
        const int row  = tid >> 1;
        const int hsel = (tid & 1) * 64;
        const int idx  = perm[b * 128 + row];
        _Float16* dst = &xT[row * 128 + hsel];
        if (idx >= 0) {
            const v4f* p = (const v4f*)(x + (size_t)idx * DIM + hsel);
            #pragma unroll
            for (int i = 0; i < 16; ++i) {
                v4f v = __builtin_nontemporal_load(p + i);   // x streamed once: TH=NT
                dst[4*i+0] = (_Float16)v[0]; dst[4*i+1] = (_Float16)v[1];
                dst[4*i+2] = (_Float16)v[2]; dst[4*i+3] = (_Float16)v[3];
            }
        } else {
            #pragma unroll
            for (int i = 0; i < 64; ++i) dst[i] = (_Float16)0.f;
        }
    }
    asm volatile("s_wait_asynccnt 0x0" ::: "memory");   // weight LDS writes complete
    __syncthreads();

    const int w    = tid >> 5;          // wave id 0..7 (wave32)
    const int l    = tid & 31;          // lane
    const int lrow = l & 15;            // A row / D column within tile
    const int lhi  = l >> 4;            // lane-half selector
    const int R    = w * 16;            // wave's M strip

    v8f acc[8];

    // ================= layer 1: h1 = relu(x @ W1[s] + b1[s]) =================
    #pragma unroll
    for (int nt = 0; nt < 8; ++nt)
        #pragma unroll
        for (int e = 0; e < 8; ++e) acc[nt][e] = 0.0f;

    #pragma unroll
    for (int ks = 0; ks < 4; ++ks) {
        // A fragment: lane holds row (R+lrow); halves 0..7 -> K = 32ks+8lhi.., halves 8..15 -> +16
        const _Float16* ap = &xT[(R + lrow) * 128 + ks * 32 + lhi * 8];
        v8h a0 = *(const v8h*)ap;
        v8h a1 = *(const v8h*)(ap + 16);
        v16h afrag = __builtin_shufflevector(a0, a1, 0,1,2,3,4,5,6,7,8,9,10,11,12,13,14,15);
        #pragma unroll
        for (int nt = 0; nt < 8; ++nt) {
            v16h bfrag = *(const v16h*)(&wL1[((ks * 8 + nt) * 32 + l) * 16]);
            acc[nt] = __builtin_amdgcn_wmma_f32_16x16x32_f16(
                false, afrag, false, bfrag, (short)0, acc[nt], false, false);
        }
    }
    #pragma unroll
    for (int nt = 0; nt < 8; ++nt) {
        const int col = nt * 16 + lrow;
        const float bv = b1[s * DIM + col];
        #pragma unroll
        for (int j = 0; j < 8; ++j) {
            float v = acc[nt][j] + bv;
            v = v > 0.f ? v : 0.f;
            hT[(R + lhi * 8 + j) * 128 + col] = (_Float16)v;   // row-major [M][K]
        }
    }
    // wave reads back only its own rows of hT -> intra-wave DS ordering suffices, no barrier

    // ================= layer 2: h2 = relu(h1 @ W2[s] + b2[s]); e = h2 . W3[s] =================
    #pragma unroll
    for (int nt = 0; nt < 8; ++nt)
        #pragma unroll
        for (int e = 0; e < 8; ++e) acc[nt][e] = 0.0f;

    #pragma unroll
    for (int ks = 0; ks < 4; ++ks) {
        const _Float16* ap = &hT[(R + lrow) * 128 + ks * 32 + lhi * 8];
        v8h a0 = *(const v8h*)ap;
        v8h a1 = *(const v8h*)(ap + 16);
        v16h afrag = __builtin_shufflevector(a0, a1, 0,1,2,3,4,5,6,7,8,9,10,11,12,13,14,15);
        #pragma unroll
        for (int nt = 0; nt < 8; ++nt) {
            v16h bfrag = *(const v16h*)(&wL2[((ks * 8 + nt) * 32 + l) * 16]);
            acc[nt] = __builtin_amdgcn_wmma_f32_16x16x32_f16(
                false, afrag, false, bfrag, (short)0, acc[nt], false, false);
        }
    }
    float partial[8];
    #pragma unroll
    for (int j = 0; j < 8; ++j) partial[j] = 0.f;
    #pragma unroll
    for (int nt = 0; nt < 8; ++nt) {
        const int col  = nt * 16 + lrow;
        const float bv  = b2[s * DIM + col];
        const float w3v = W3[s * DIM + col];
        #pragma unroll
        for (int j = 0; j < 8; ++j) {
            float v = acc[nt][j] + bv;
            v = v > 0.f ? v : 0.f;
            partial[j] += v * w3v;
        }
    }
    // reduce over N: butterfly within each 16-lane half (lanes 0-15 -> M 0..7, 16-31 -> M 8..15)
    #pragma unroll
    for (int j = 0; j < 8; ++j) {
        float v = partial[j];
        v += __shfl_xor(v, 8, 32);
        v += __shfl_xor(v, 4, 32);
        v += __shfl_xor(v, 2, 32);
        v += __shfl_xor(v, 1, 32);
        partial[j] = v;
    }
    if (lrow == 0) {
        const float sl = slope[s], ic = intercept[s], bb = b3[s];
        #pragma unroll
        for (int j = 0; j < 8; ++j) {
            const int m = R + lhi * 8 + j;
            const int pidx = perm[b * 128 + m];
            if (pidx >= 0) e_atom[pidx] = sl * (partial[j] + bb) + ic;
        }
    }
}

// ---------------- deterministic per-image segment sum ----------------

__global__ __launch_bounds__(256) void k_segsum(const int* __restrict__ image_ids,
                                                const float* __restrict__ e_atom,
                                                float* __restrict__ out, int n) {
    __shared__ float red[256];
    const int b = blockIdx.x;
    // lower_bound(b), lower_bound(b+1) on sorted image_ids
    int lo = 0, hi = n;
    while (lo < hi) { int mid = (lo + hi) >> 1; if (image_ids[mid] < b) lo = mid + 1; else hi = mid; }
    int lo2 = lo, hi2 = n;
    while (lo2 < hi2) { int mid = (lo2 + hi2) >> 1; if (image_ids[mid] < b + 1) lo2 = mid + 1; else hi2 = mid; }
    float acc = 0.f;
    for (int i = lo + threadIdx.x; i < lo2; i += 256) acc += e_atom[i];
    red[threadIdx.x] = acc;
    __syncthreads();
    for (int st = 128; st > 0; st >>= 1) {
        if ((int)threadIdx.x < st) red[threadIdx.x] += red[threadIdx.x + st];
        __syncthreads();
    }
    if (threadIdx.x == 0) out[b] = red[0];
}

// ---------------- launch ----------------

extern "C" void kernel_launch(void* const* d_in, const int* in_sizes, int n_in,
                              void* d_out, int out_size, void* d_ws, size_t ws_size,
                              hipStream_t stream) {
    const float* x         = (const float*)d_in[0];
    const float* W1        = (const float*)d_in[1];
    const float* b1        = (const float*)d_in[2];
    const float* W2        = (const float*)d_in[3];
    const float* b2        = (const float*)d_in[4];
    const float* W3        = (const float*)d_in[5];
    const float* b3        = (const float*)d_in[6];
    const float* slope     = (const float*)d_in[7];
    const float* intercept = (const float*)d_in[8];
    const int*   sym       = (const int*)d_in[9];
    const int*   img       = (const int*)d_in[10];
    float*       out       = (float*)d_out;

    const int n = in_sizes[9];               // N atoms
    if (ws_size < (size_t)WS_NEEDED) return; // scratch too small (deterministic no-op)

    char* ws = (char*)d_ws;
    int*      cnt       = (int*)(ws + OFF_CNT);
    int*      cur       = (int*)(ws + OFF_CUR);
    int*      tileStart = (int*)(ws + OFF_TS);
    int*      rowBase   = (int*)(ws + OFF_RB);
    int*      perm      = (int*)(ws + OFF_PERM);
    float*    e_atom    = (float*)(ws + OFF_EATOM);
    _Float16* pW1       = (_Float16*)(ws + OFF_PW1);
    _Float16* pW2       = (_Float16*)(ws + OFF_PW2);

    const int nb256 = (n + 255) / 256;
    k_zero_counters<<<1, 32, 0, stream>>>((int*)ws);
    k_hist<<<nb256, 256, 0, stream>>>(sym, cnt, n);
    k_setup_offsets<<<1, 1, 0, stream>>>(cnt, tileStart, rowBase, perm);
    k_scatter<<<nb256, 256, 0, stream>>>(sym, rowBase, cur, perm, n);
    k_pack_weights<<<512, 256, 0, stream>>>(W1, W2, pW1, pW2);

    const int max_tiles = n / 128 + NSYM;    // padded tile upper bound
    k_moe_wmma<<<max_tiles, 256, 0, stream>>>(x, b1, b2, W3, b3, slope, intercept,
                                              pW1, pW2, perm, tileStart, e_atom);
    k_segsum<<<out_size, 256, 0, stream>>>(img, e_atom, out, n);
}
